// Mamba2Block_8134668059074
// MI455X (gfx1250) — compile-verified, matrix operand order checked by reading, not run on hardware
//
#include <hip/hip_runtime.h>
#include <hip/hip_bf16.h>
#include <stdint.h>

// ---------------- problem constants (from reference) ----------------
#define BATCH    2
#define SEQLEN   2048
#define DMODEL   1024
#define DINNER   2048
#define NHEADS   8
#define HEADDIM  256
#define DSTATE   16
#define DXBC     2080            // D_INNER + 2*NGROUPS*D_STATE
#define LDZ      8232            // D_IN_PROJ
#define MTOT     (BATCH*SEQLEN)  // 4096 token rows

// ---------------- vector types ----------------
typedef __attribute__((ext_vector_type(16))) __bf16        v16bf;
typedef __attribute__((ext_vector_type(8)))  float         v8f;
typedef __attribute__((ext_vector_type(4)))  unsigned int  u32x4;
typedef __attribute__((ext_vector_type(8)))  int           i32x8;
typedef __attribute__((ext_vector_type(4)))  int           i32x4;

// ---------------- GEMM tiling ----------------
#define BM 128
#define BN 128
#define BK 32

__device__ __forceinline__ float sigmoidf_(float x) { return 1.0f / (1.0f + __expf(-x)); }

// ---------------- Tensor Data Mover: 2-D tile -> LDS ----------------
#if __has_builtin(__builtin_amdgcn_tensor_load_to_lds)
#define HAVE_TDM 1
#else
#define HAVE_TDM 0
#endif

#if HAVE_TDM
// Pack D# group0/group1 per CDNA5 ISA §8.3/§8.4 for a 2-D bf16 tile load.
// Tensor dims are "remaining" extents from the tile origin so the TDM's OOB
// logic zero-fills the ragged edge (N=8232 is not a multiple of BN).
__device__ __forceinline__ void tdm_load_2d(unsigned lds_addr, const void* gptr,
                                            unsigned td0, unsigned td1,
                                            unsigned tile0, unsigned tile1,
                                            unsigned long long stride0_elems)
{
  unsigned long long ga = (unsigned long long)(uintptr_t)gptr;
  u32x4 g0 = { 1u,                                       // count=1 (valid user D#)
               lds_addr,                                 // lds_addr[31:0]
               (unsigned)ga,                             // global_addr[31:0]
               (unsigned)((ga >> 32) & 0x01FFFFFFull)    // global_addr[56:32]
                 | 0x80000000u };                        // type=2 ("image") in [127:126]
  i32x8 g1 = { (int)(1u << 16),                          // data_size=1 (2 bytes/elem)
               (int)((td0 & 0xFFFFu) << 16),             // tensor_dim0[15:0]
               (int)((td0 >> 16) | ((td1 & 0xFFFFu) << 16)),
               (int)((td1 >> 16) | (tile0 << 16)),       // tile_dim0
               (int)(tile1 & 0xFFFFu),                   // tile_dim1 (tile_dim2=0 -> 2D)
               (int)(stride0_elems & 0xFFFFFFFFull),     // tensor_dim0_stride[31:0]
               (int)((stride0_elems >> 32) & 0xFFFFull), // stride[47:32]
               0 };
  i32x4 gz = {0, 0, 0, 0};
#if defined(__clang_major__) && (__clang_major__ >= 23)
  i32x8 gz8 = {0, 0, 0, 0, 0, 0, 0, 0};
  __builtin_amdgcn_tensor_load_to_lds(g0, g1, gz, gz, gz8, 0);
#else
  __builtin_amdgcn_tensor_load_to_lds(g0, g1, gz, gz, 0);
#endif
}
#endif

// ---------------- WMMA fragment load from LDS ----------------
// 16-bit A/B layout (ISA §7.12.2): lane<16 row r holds K {0..7, 16..23},
// lane>=16 holds K {8..15, 24..31} -> two ds_load_b128 per fragment.
__device__ __forceinline__ v16bf load_frag16(const __bf16* s, int row0, int lane)
{
  const int r  = lane & 15;
  const int hl = lane >> 4;
  const __bf16* base = s + (row0 + r) * BK;
  union { v16bf v; float4 f[2]; } u;
  u.f[0] = *reinterpret_cast<const float4*>(base + hl * 8);
  u.f[1] = *reinterpret_cast<const float4*>(base + 16 + hl * 8);
  return u.v;
}

// ---------------- WMMA GEMM: C[M,ldc] = A[M,K] * Bw[N,K]^T ----------------
// 8 waves / block; block tile 128x128; each wave owns 32x64 (2x4 WMMA tiles).
// Double-buffered LDS filled by the Tensor Data Mover: wave0 keeps up to
// 4 tile-loads in flight; s_wait_tensorcnt(2) gates the current buffer
// (in-order completion => trailing 2 ops belong to the next buffer).
__global__ __launch_bounds__(256) void wmma_gemm_bf16(
    const __bf16* __restrict__ A,   // [M,K] row-major
    const __bf16* __restrict__ Bw,  // [N,K] row-major (weights; acts as B^T)
    float* __restrict__ C,          // [M,ldc] f32
    int M, int N, int K, int ldc)
{
  __shared__ __align__(16) __bf16 sA[2][BM * BK];  // 2 x 8 KB
  __shared__ __align__(16) __bf16 sB[2][BN * BK];  // 2 x 8 KB

  const int m0   = blockIdx.y * BM;
  const int n0   = blockIdx.x * BN;
  const int tid  = threadIdx.x;
  const int wid  = tid >> 5;        // wave32
  const int lane = tid & 31;
  const int wm0  = (wid >> 1) * 32; // 4 wave-rows of 32
  const int wn0  = (wid & 1) * 64;  // 2 wave-cols of 64

  const v8f vz = {0, 0, 0, 0, 0, 0, 0, 0};
  v8f acc[2][4];
#pragma unroll
  for (int i = 0; i < 2; ++i)
#pragma unroll
    for (int j = 0; j < 4; ++j) acc[i][j] = vz;

  const int nk = K / BK;

#if HAVE_TDM
  auto issue_tiles = [&](int kt, int buf) {
    const int kk = kt * BK;
    tdm_load_2d((unsigned)(uintptr_t)&sA[buf][0], A + (size_t)m0 * K + kk,
                (unsigned)(K - kk), (unsigned)(M - m0), BK, BM,
                (unsigned long long)K);
    tdm_load_2d((unsigned)(uintptr_t)&sB[buf][0], Bw + (size_t)n0 * K + kk,
                (unsigned)(K - kk), (unsigned)(N - n0), BK, BN,
                (unsigned long long)K);
  };
  if (wid == 0) {
    issue_tiles(0, 0);
    if (nk > 1) issue_tiles(1, 1);
  }
#endif

  for (int kt = 0; kt < nk; ++kt) {
    const int cur = kt & 1;
#if HAVE_TDM
    if (wid == 0) {
      if (kt + 1 < nk) __builtin_amdgcn_s_wait_tensorcnt(2);  // current buf done
      else             __builtin_amdgcn_s_wait_tensorcnt(0);  // drain at tail
    }
#else
    {
      const int k0 = kt * BK;
      for (int c = tid; c < (BM * BK) / 8; c += 256) {
        const int row = c >> 2, kc = (c & 3) * 8;
        *reinterpret_cast<float4*>(&sA[cur][row * BK + kc]) =
            *reinterpret_cast<const float4*>(A + (size_t)(m0 + row) * K + k0 + kc);
      }
      for (int c = tid; c < (BN * BK) / 8; c += 256) {
        const int row = c >> 2, kc = (c & 3) * 8;
        float4 v = {0.f, 0.f, 0.f, 0.f};
        if (n0 + row < N)
          v = *reinterpret_cast<const float4*>(Bw + (size_t)(n0 + row) * K + k0 + kc);
        *reinterpret_cast<float4*>(&sB[cur][row * BK + kc]) = v;
      }
    }
#endif
    __syncthreads();

    v16bf afrag[2], bfrag[4];
#pragma unroll
    for (int i = 0; i < 2; ++i) afrag[i] = load_frag16(&sA[cur][0], wm0 + 16 * i, lane);
#pragma unroll
    for (int j = 0; j < 4; ++j) bfrag[j] = load_frag16(&sB[cur][0], wn0 + 16 * j, lane);

#pragma unroll
    for (int i = 0; i < 2; ++i)
#pragma unroll
      for (int j = 0; j < 4; ++j)
        acc[i][j] = __builtin_amdgcn_wmma_f32_16x16x32_bf16(
            false, afrag[i], false, bfrag[j], (short)0, acc[i][j], false, false);

    __syncthreads();   // all waves done reading before buf `cur` is refilled

#if HAVE_TDM
    if (wid == 0 && kt + 2 < nk) issue_tiles(kt + 2, cur);  // overlaps compute of kt+1
#endif
  }

  // C/D layout: VGPR v -> M = v (lanes 0-15) / 8+v (lanes 16-31); N = lane&15
  const int r = lane & 15, hl = lane >> 4;
#pragma unroll
  for (int i = 0; i < 2; ++i) {
#pragma unroll
    for (int j = 0; j < 4; ++j) {
      const int nn = n0 + wn0 + 16 * j + r;
      if (nn < N) {
        const int mbase = m0 + wm0 + 16 * i + 8 * hl;
#pragma unroll
        for (int v = 0; v < 8; ++v)
          C[(size_t)(mbase + v) * ldc + nn] = acc[i][j][v];
      }
    }
  }
}

// ---------------- f32 -> bf16 cast ----------------
__global__ __launch_bounds__(256) void cast_bf16_kernel(
    const float* __restrict__ in, __bf16* __restrict__ out, long n)
{
  const long i = (long)blockIdx.x * 256 + threadIdx.x;
  if (i < n) out[i] = (__bf16)in[i];
}

// ---------------- dt = softplus(zx[:, -8:] + dt_bias) ----------------
__global__ __launch_bounds__(256) void dt_softplus_kernel(
    const float* __restrict__ zx, const float* __restrict__ dt_bias,
    float* __restrict__ dt_out)
{
  const int i = blockIdx.x * 256 + threadIdx.x;
  if (i >= MTOT * NHEADS) return;
  const int r = i >> 3, h = i & 7;
  const float v = zx[(size_t)r * LDZ + (LDZ - NHEADS) + h] + dt_bias[h];
  dt_out[i] = (v > 20.f) ? v : log1pf(__expf(v));
}

// ---------------- causal depthwise conv(4) + bias + SiLU ----------------
__global__ __launch_bounds__(256) void conv_silu_kernel(
    const float* __restrict__ zx, const float* __restrict__ w,
    const float* __restrict__ bias, float* __restrict__ out)
{
  const long i = (long)blockIdx.x * 256 + threadIdx.x;
  const long total = (long)MTOT * DXBC;
  if (i >= total) return;
  const int  c  = (int)(i % DXBC);
  const long bl = i / DXBC;
  const int  l  = (int)(bl % SEQLEN);
  const long bb = bl / SEQLEN;
  float acc = bias[c];
#pragma unroll
  for (int j = 0; j < 4; ++j) {
    const int ls = l - 3 + j;
    if (ls >= 0)
      acc += w[c * 4 + j] * zx[(size_t)(bb * SEQLEN + ls) * LDZ + 3 * DINNER + c];
  }
  out[i] = acc * sigmoidf_(acc);
}

// ---------------- sequential SSM scan ----------------
// one workgroup per (batch, head); each thread owns one HEADDIM slot and
// keeps its 16-state row in registers for the whole sequence.
__global__ __launch_bounds__(HEADDIM) void ssm_scan_kernel(
    const float* __restrict__ xbc,    // [MTOT, DXBC] post-conv
    const float* __restrict__ dt,     // [MTOT, NHEADS]
    const float* __restrict__ A_log,
    const float* __restrict__ D_param,
    float* __restrict__ y)            // [MTOT, DINNER]
{
  const int b = blockIdx.x / NHEADS;
  const int h = blockIdx.x % NHEADS;
  const int p = threadIdx.x;
  const float Aneg = -__expf(A_log[h]);
  const float Dp   = D_param[h];
  float hs[DSTATE];
#pragma unroll
  for (int n = 0; n < DSTATE; ++n) hs[n] = 0.f;

  __shared__ float sB[DSTATE], sC[DSTATE], sdt[1];

  for (int l = 0; l < SEQLEN; ++l) {
    const size_t bl = (size_t)b * SEQLEN + l;
    const float* row = xbc + bl * DXBC;
    if (p < DSTATE)             sB[p]          = row[DINNER + p];
    else if (p < 2 * DSTATE)    sC[p - DSTATE] = row[DINNER + p];
    if (p == 2 * DSTATE)        sdt[0]         = dt[bl * NHEADS + h];
    __syncthreads();

    const float x_t = row[h * HEADDIM + p];
    if (l + 1 < SEQLEN)
      __builtin_prefetch(row + DXBC + h * HEADDIM + p, 0, 1);  // global_prefetch next step

    const float dtv = sdt[0];
    const float dA  = __expf(dtv * Aneg);
    const float dbx = dtv * x_t;
    float yv = 0.f;
#pragma unroll
    for (int n = 0; n < DSTATE; ++n) {
      hs[n] = dA * hs[n] + dbx * sB[n];
      yv += hs[n] * sC[n];
    }
    y[bl * DINNER + h * HEADDIM + p] = yv + Dp * x_t;
    __syncthreads();
  }
}

// ---------------- LayerNorm over DINNER; emits f32 + bf16 copies ----------------
__global__ __launch_bounds__(256) void layernorm_kernel(
    const float* __restrict__ y, const float* __restrict__ gamma,
    const float* __restrict__ beta,
    float* __restrict__ y_ln, __bf16* __restrict__ y_bf)
{
  const int row = blockIdx.x;
  const float* yr = y + (size_t)row * DINNER;
  float s = 0.f, ss = 0.f;
  for (int i = threadIdx.x; i < DINNER; i += 256) {
    const float v = yr[i];
    s += v; ss += v * v;
  }
  __shared__ float rs[256], rq[256];
  rs[threadIdx.x] = s; rq[threadIdx.x] = ss;
  __syncthreads();
  for (int o = 128; o > 0; o >>= 1) {
    if (threadIdx.x < o) { rs[threadIdx.x] += rs[threadIdx.x + o]; rq[threadIdx.x] += rq[threadIdx.x + o]; }
    __syncthreads();
  }
  const float mu  = rs[0] * (1.f / DINNER);
  const float var = rq[0] * (1.f / DINNER) - mu * mu;
  const float inv = rsqrtf(var + 1e-5f);
  for (int i = threadIdx.x; i < DINNER; i += 256) {
    const float v = (yr[i] - mu) * inv * gamma[i] + beta[i];
    y_ln[(size_t)row * DINNER + i] = v;
    y_bf[(size_t)row * DINNER + i] = (__bf16)v;
  }
}

// ---------------- gate*y and silu(z0)*x0 -> concatenated bf16 ----------------
__global__ __launch_bounds__(256) void combine_kernel(
    const float* __restrict__ zx, const float* __restrict__ gate_pre,
    const float* __restrict__ b_gate, const float* __restrict__ y_ln,
    __bf16* __restrict__ cat)
{
  const long i = (long)blockIdx.x * 256 + threadIdx.x;
  const long total = (long)MTOT * (2 * DINNER);
  if (i >= total) return;
  const long row = i / (2 * DINNER);
  const int  c   = (int)(i % (2 * DINNER));
  float val;
  if (c < DINNER) {
    const float z0 = zx[row * (size_t)LDZ + c];
    const float x0 = zx[row * (size_t)LDZ + DINNER + c];
    val = z0 * sigmoidf_(z0) * x0;                 // silu(z0)*x0
  } else {
    const int k = c - DINNER;
    const float g = sigmoidf_(gate_pre[row * (size_t)DINNER + k] + b_gate[k]);
    val = g * y_ln[row * (size_t)DINNER + k];
  }
  cat[i] = (__bf16)val;
}

// ---------------- launch ----------------
extern "C" void kernel_launch(void* const* d_in, const int* in_sizes, int n_in,
                              void* d_out, int out_size, void* d_ws, size_t ws_size,
                              hipStream_t stream)
{
  const float* u        = (const float*)d_in[0];
  const float* W_in     = (const float*)d_in[1];
  const float* conv_w   = (const float*)d_in[2];
  const float* conv_b   = (const float*)d_in[3];
  const float* dt_bias  = (const float*)d_in[4];
  const float* A_log    = (const float*)d_in[5];
  const float* D_param  = (const float*)d_in[6];
  const float* ln_gamma = (const float*)d_in[7];
  const float* ln_beta  = (const float*)d_in[8];
  const float* W_gate   = (const float*)d_in[9];
  const float* b_gate   = (const float*)d_in[10];
  const float* W_out    = (const float*)d_in[11];
  (void)in_sizes; (void)n_in; (void)out_size; (void)ws_size;

  char* ws = (char*)d_ws;
  size_t off = 0;
  auto take = [&](size_t bytes) -> char* {
    char* p = ws + off;
    off = (off + bytes + 255) & ~(size_t)255;
    return p;
  };
  float*  zx     = (float*) take((size_t)MTOT * LDZ * 4);       // in_proj output
  __bf16* u_bf   = (__bf16*)take((size_t)MTOT * DMODEL * 2);
  __bf16* win_bf = (__bf16*)take((size_t)LDZ * DMODEL * 2);
  float*  xbc    = (float*) take((size_t)MTOT * DXBC * 4);      // conv+silu output
  float*  dtb    = (float*) take((size_t)MTOT * NHEADS * 4);
  float*  yraw   = (float*) take((size_t)MTOT * DINNER * 4);    // scan output
  float*  yln    = (float*) take((size_t)MTOT * DINNER * 4);
  __bf16* y_bf   = (__bf16*)take((size_t)MTOT * DINNER * 2);
  __bf16* wg_bf  = (__bf16*)take((size_t)DINNER * DINNER * 2);
  float*  gpre   = (float*) take((size_t)MTOT * DINNER * 4);
  __bf16* cat_bf = (__bf16*)take((size_t)MTOT * 2 * DINNER * 2);
  __bf16* wo_bf  = (__bf16*)take((size_t)DMODEL * 2 * DINNER * 2);

  auto castb = [&](const float* in, __bf16* o, long n) {
    cast_bf16_kernel<<<dim3((unsigned)((n + 255) / 256)), dim3(256), 0, stream>>>(in, o, n);
  };
  castb(u,      u_bf,   (long)MTOT * DMODEL);
  castb(W_in,   win_bf, (long)LDZ * DMODEL);
  castb(W_gate, wg_bf,  (long)DINNER * DINNER);
  castb(W_out,  wo_bf,  (long)DMODEL * 2 * DINNER);

  // 1) in_proj: zx[4096,8232] = u * W_in^T
  wmma_gemm_bf16<<<dim3((LDZ + BN - 1) / BN, MTOT / BM), 256, 0, stream>>>(
      u_bf, win_bf, zx, MTOT, LDZ, DMODEL, LDZ);

  // 2) dt + conv/SiLU
  dt_softplus_kernel<<<dim3((MTOT * NHEADS) / 256), 256, 0, stream>>>(zx, dt_bias, dtb);
  conv_silu_kernel<<<dim3((unsigned)(((long)MTOT * DXBC + 255) / 256)), 256, 0, stream>>>(
      zx, conv_w, conv_b, xbc);

  // 3) sequential state-space scan (16 persistent workgroups)
  ssm_scan_kernel<<<dim3(BATCH * NHEADS), dim3(HEADDIM), 0, stream>>>(
      xbc, dtb, A_log, D_param, yraw);

  // 4) LayerNorm
  layernorm_kernel<<<dim3(MTOT), 256, 0, stream>>>(yraw, ln_gamma, ln_beta, yln, y_bf);

  // 5) gate GEMM: gpre[4096,2048] = y_ln * W_gate^T
  wmma_gemm_bf16<<<dim3(DINNER / BN, MTOT / BM), 256, 0, stream>>>(
      y_bf, wg_bf, gpre, MTOT, DINNER, DINNER, DINNER);

  // 6) sigmoid-gate * y  and  silu(z0)*x0  -> cat (bf16)
  combine_kernel<<<dim3((unsigned)(((long)MTOT * 2 * DINNER) / 256)), 256, 0, stream>>>(
      zx, gpre, b_gate, yln, cat_bf);

  // 7) out_proj: out[4096,1024] = cat * W_out^T (f32 output)
  wmma_gemm_bf16<<<dim3(DMODEL / BN, MTOT / BM), 256, 0, stream>>>(
      cat_bf, wo_bf, (float*)d_out, MTOT, DMODEL, 2 * DINNER, DMODEL);
}
